// BlockResMLP_MixerBlock_47888885350878
// MI455X (gfx1250) — matrix operand order repaired and checked by reading, not compiled, not run on hardware
//
#include <hip/hip_runtime.h>

// ---------------- problem constants ----------------
#define BLOCKD 64
#define HID    128
#define IN_DIM 4096
#define BS     8192
#define NBLK   64
#define MT     128          // rows per workgroup

// ---------------- LDS layout (bytes) ----------------
// zf  : float[128][68]   (z tile fp32, +4 float pad = TDM pad 64dw/4dw) -> 34816
// hb  : bf16 [128][136]  (hidden bf16, +8 bf16 pad); reused as fp32
//                        staging float[128][64] (32768B) for the store  -> 34816
// w1t : bf16 [128][72]   (W1^T: N=128 rows, K=64+8)                     -> 18432
// w2t : bf16 [64][136]   (W2^T: N=64 rows, K=128+8)                     -> 17408
// b1s : float[128], b2s : float[64]
#define ZSTR 68
#define HSTR 136
#define W1STR 72
#define W2STR 136
#define OFF_ZF  0
#define OFF_HB  34816
#define OFF_W1T 69632
#define OFF_W2T 88064
#define OFF_B1  105472
#define OFF_B2  105984
#define SMEM_BYTES 106240

typedef __attribute__((ext_vector_type(16))) __bf16 v16bf;
typedef __attribute__((ext_vector_type(8)))  __bf16 v8bf;
typedef __attribute__((ext_vector_type(8)))  float  v8f;
typedef __attribute__((ext_vector_type(4)))  float  v4f;
typedef __attribute__((ext_vector_type(4)))  unsigned v4u;
typedef __attribute__((ext_vector_type(8)))  int      v8i;
typedef __attribute__((ext_vector_type(4)))  int      v4i;

#if __has_builtin(__builtin_amdgcn_tensor_load_to_lds) && \
    __has_builtin(__builtin_amdgcn_tensor_store_from_lds) && \
    __has_builtin(__builtin_amdgcn_s_wait_tensorcnt)
#define USE_TDM 1
#else
#define USE_TDM 0
#endif

__device__ __forceinline__ float elu1(float x) {
    return x > 0.f ? x : (__expf(x) - 1.f);
}

__device__ __forceinline__ v16bf cat8(v8bf lo, v8bf hi) {
    return __builtin_shufflevector(lo, hi, 0,1,2,3,4,5,6,7,8,9,10,11,12,13,14,15);
}

#if USE_TDM
// One TDM transfer of the (128 x 64) activation tile.
//  layer 1 (strided==0): 2D tile, dim0 = 64 contiguous f32, dim1 = 128 rows @ stride 4096
//  layer 2 (strided==1): 3D tile, dim0 = 1,  dim1 = 64 @ stride 64, dim2 = 128 @ stride 4096
//  use_pad: insert 4 dwords of LDS padding every 64 dwords (matches ZSTR=68), load only.
__device__ __forceinline__ void tdm_tile(bool is_store, unsigned lds_off,
                                         const float* gbase, int strided, int use_pad) {
    unsigned long long ga = (unsigned long long)(const void*)gbase;
    v4u g0;
    g0[0] = 1u;                                         // count=1 user descriptor
    g0[1] = lds_off;                                    // LDS byte address (dyn-LDS base = 0)
    g0[2] = (unsigned)ga;                               // global addr [31:0]
    g0[3] = (unsigned)((ga >> 32) & 0x1FFFFFFu) | (2u << 30);  // addr[56:32] | type=2

    unsigned w0 = (2u << 16);                           // data_size = 4 bytes
    if (use_pad) w0 |= (1u << 20) | (5u << 22) | (3u << 25);   // pad: every 64dw add 4dw
    const unsigned TD = 0x00FFFFFFu;                    // huge tensor dims: no OOB clamp
    unsigned tile0 = strided ? 1u : 64u;
    unsigned tile1 = strided ? 64u : 128u;
    unsigned tile2 = strided ? 128u : 0u;
    unsigned s0    = strided ? 64u : 4096u;             // tensor_dim0_stride (elements)
    unsigned s1    = strided ? 4096u : 0u;              // tensor_dim1_stride (elements)
    v8i g1;
    g1[0] = (int)w0;
    g1[1] = (int)((TD & 0xFFFFu) << 16);                          // tensor_dim0 lo16
    g1[2] = (int)(((TD >> 16) & 0xFFFFu) | ((TD & 0xFFFFu) << 16));// dim0 hi | dim1 lo
    g1[3] = (int)(((TD >> 16) & 0xFFFFu) | (tile0 << 16));         // dim1 hi | tile_dim0
    g1[4] = (int)(tile1 | (tile2 << 16));                          // tile_dim1 | tile_dim2
    g1[5] = (int)s0;                                               // stride0 [31:0]
    g1[6] = (int)((s1 & 0xFFFFu) << 16);                           // stride0 hi=0 | stride1 lo16
    g1[7] = 0;                                                     // stride1 hi
    v4i g2; g2[0] = (int)TD; g2[1] = 0; g2[2] = 0; g2[3] = 0;      // tensor_dim2; tile_dim3=0
    v4i g3; g3[0] = 0; g3[1] = 0; g3[2] = 0; g3[3] = 0;
    v8i g4; g4[0] = 0; g4[1] = 0; g4[2] = 0; g4[3] = 0;            // extra group (clang-23
    g4[4] = 0; g4[5] = 0; g4[6] = 0; g4[7] = 0;                    // 6-arg form), unused
    if (is_store) __builtin_amdgcn_tensor_store_from_lds(g0, g1, g2, g3, g4, 0);
    else          __builtin_amdgcn_tensor_load_to_lds(g0, g1, g2, g3, g4, 0);
}
#endif

// One mixer layer per (block n, 128-row tile) workgroup.
// feature(row b, channel c, block n) = n*sn + c*sd
// layer 0: sd=1, sn=64 ; layer 1 (transposed shuffle): sd=64, sn=1
__global__ __launch_bounds__(256, 2)
void mixer_layer_wmma(const float* __restrict__ in, float* __restrict__ out,
                      const float* __restrict__ w1, const float* __restrict__ b1,
                      const float* __restrict__ w2, const float* __restrict__ b2,
                      int sd, int sn)
{
    extern __shared__ char smem[];
    float*  zf  = (float*) (smem + OFF_ZF);
    __bf16* hb  = (__bf16*)(smem + OFF_HB);
    float*  stg = (float*) (smem + OFF_HB);   // reused after hb is consumed
    __bf16* w1t = (__bf16*)(smem + OFF_W1T);
    __bf16* w2t = (__bf16*)(smem + OFF_W2T);
    float*  b1s = (float*) (smem + OFF_B1);
    float*  b2s = (float*) (smem + OFF_B2);

    const int tid  = threadIdx.x;
    const int nblk = blockIdx.x;
    const int row0 = blockIdx.y * MT;
    const size_t featbase = (size_t)nblk * sn;

    // ---- async: Tensor Data Mover pulls the z tile into LDS (with padding) ----
#if USE_TDM
    if (tid < 32) {
        tdm_tile(false, OFF_ZF, in + (size_t)row0 * IN_DIM + featbase, sd == 64, 1);
    }
#else
    #pragma unroll 4
    for (int i = tid; i < MT * BLOCKD; i += 256) {
        int r = i >> 6, c = i & 63;
        zf[r * ZSTR + c] = in[(size_t)(row0 + r) * IN_DIM + featbase + (size_t)c * sd];
    }
#endif

    // ---- stage weights (bf16, transposed: row-major N x K, operand friendly) ----
    const float* w1g = w1 + (size_t)nblk * (BLOCKD * HID);   // [64][128]
    const float* w2g = w2 + (size_t)nblk * (HID * BLOCKD);   // [128][64]
    #pragma unroll 4
    for (int i = tid; i < BLOCKD * HID; i += 256) {
        int d = i >> 7, e = i & 127;
        w1t[e * W1STR + d] = (__bf16)w1g[i];
    }
    #pragma unroll 4
    for (int i = tid; i < HID * BLOCKD; i += 256) {
        int e = i >> 6, d = i & 63;
        w2t[d * W2STR + e] = (__bf16)w2g[i];
    }
    if (tid < HID)    b1s[tid] = b1[(size_t)nblk * HID + tid];
    if (tid < BLOCKD) b2s[tid] = b2[(size_t)nblk * BLOCKD + tid];

#if USE_TDM
    if (tid < 32) __builtin_amdgcn_s_wait_tensorcnt(0);
#endif
    __syncthreads();

    const int wave = tid >> 5;
    const int lane = tid & 31;
    const int lm   = lane & 15;      // M (A) or N (B/D) within tile
    const int lh   = lane >> 4;      // half-select

    // ================= GEMM1: h = ELU(z(128x64) * W1(64x128) + b1) =================
    // wave w owns M-tile w (16 rows) x all 8 N-tiles; K=64 -> 2 wmma steps.
    {
        const int mt = wave;
        const v4f* zp = (const v4f*)&zf[(mt * 16 + lm) * ZSTR];
        // A 16-bit layout: lanes0-15 K={kb..kb+7, kb+16..kb+23}; lanes16-31 shift K by 8
        v4f u0 = zp[0 + lh * 2], u1 = zp[1 + lh * 2];
        v4f u2 = zp[4 + lh * 2], u3 = zp[5 + lh * 2];
        v4f u4 = zp[8 + lh * 2], u5 = zp[9 + lh * 2];
        v4f u6 = zp[12 + lh * 2], u7 = zp[13 + lh * 2];
        v16bf a0, a1;
        #pragma unroll
        for (int i = 0; i < 4; ++i) {
            a0[i]      = (__bf16)u0[i];  a0[4 + i]  = (__bf16)u1[i];
            a0[8 + i]  = (__bf16)u2[i];  a0[12 + i] = (__bf16)u3[i];
            a1[i]      = (__bf16)u4[i];  a1[4 + i]  = (__bf16)u5[i];
            a1[8 + i]  = (__bf16)u6[i];  a1[12 + i] = (__bf16)u7[i];
        }
        #pragma unroll
        for (int nt = 0; nt < 8; ++nt) {
            const __bf16* brow = &w1t[(nt * 16 + lm) * W1STR];
            // B 16-bit layout: lanes0-15 K={kb..kb+7, kb+16..kb+23}; lanes16-31 shift K by 8
            v16bf b0  = cat8(*(const v8bf*)(brow + lh * 8),      *(const v8bf*)(brow + 16 + lh * 8));
            v16bf b1v = cat8(*(const v8bf*)(brow + 32 + lh * 8), *(const v8bf*)(brow + 48 + lh * 8));
            v8f acc = {};
            acc = __builtin_amdgcn_wmma_f32_16x16x32_bf16(false, a0, false, b0,  (short)0, acc, false, false);
            acc = __builtin_amdgcn_wmma_f32_16x16x32_bf16(false, a1, false, b1v, (short)0, acc, false, false);
            const int col   = nt * 16 + lm;
            const float bia = b1s[col];
            const int rbase = mt * 16 + lh * 8;
            #pragma unroll
            for (int g = 0; g < 8; ++g)
                hb[(rbase + g) * HSTR + col] = (__bf16)elu1(acc[g] + bia);
        }
    }
    // NOTE: no barrier needed: each wave consumes only the h rows it produced
    // (in-wave ds ordering via s_wait_dscnt).

    // ================= GEMM2: o = h(128x128) * W2(128x64) + b2 + z =================
    float ov[4][8];
    {
        const int mt = wave;
        const __bf16* hrow = &hb[(mt * 16 + lm) * HSTR];
        v16bf ha[4];
        #pragma unroll
        for (int j = 0; j < 4; ++j)
            ha[j] = cat8(*(const v8bf*)(hrow + j * 32 + lh * 8),
                         *(const v8bf*)(hrow + j * 32 + 16 + lh * 8));
        #pragma unroll
        for (int nt = 0; nt < 4; ++nt) {
            const __bf16* wrow = &w2t[(nt * 16 + lm) * W2STR];
            v8f acc = {};
            #pragma unroll
            for (int j = 0; j < 4; ++j) {
                v16bf bv = cat8(*(const v8bf*)(wrow + j * 32 + lh * 8),
                                *(const v8bf*)(wrow + j * 32 + 16 + lh * 8));
                acc = __builtin_amdgcn_wmma_f32_16x16x32_bf16(false, ha[j], false, bv, (short)0, acc, false, false);
            }
            const int col   = nt * 16 + lm;
            const float bia = b2s[col];
            const int rbase = mt * 16 + lh * 8;
            #pragma unroll
            for (int g = 0; g < 8; ++g)
                ov[nt][g] = acc[g] + bia + zf[(rbase + g) * ZSTR + col];   // fp32 residual
        }
    }
    __syncthreads();      // all reads of hb (A-operands) and zf (residual) complete

    // stage fp32 result unpadded for the TDM store / coalesced fallback
    {
        const int rbase = wave * 16 + lh * 8;
        #pragma unroll
        for (int nt = 0; nt < 4; ++nt) {
            const int col = nt * 16 + lm;
            #pragma unroll
            for (int g = 0; g < 8; ++g)
                stg[(rbase + g) * BLOCKD + col] = ov[nt][g];
        }
    }
    __syncthreads();

#if USE_TDM
    if (tid < 32) {
        tdm_tile(true, OFF_HB, out + (size_t)row0 * IN_DIM + featbase, sd == 64, 0);
    }
    // S_ENDPGM performs an implicit wait-idle (TENSORcnt -> 0) before termination.
#else
    #pragma unroll 4
    for (int i = tid; i < MT * BLOCKD; i += 256) {
        int r = i >> 6, c = i & 63;
        out[(size_t)(row0 + r) * IN_DIM + featbase + (size_t)c * sd] = stg[r * BLOCKD + c];
    }
#endif
}

extern "C" void kernel_launch(void* const* d_in, const int* in_sizes, int n_in,
                              void* d_out, int out_size, void* d_ws, size_t ws_size,
                              hipStream_t stream) {
    const float* x  = (const float*)d_in[0];
    const float* w1 = (const float*)d_in[1];
    const float* b1 = (const float*)d_in[2];
    const float* w2 = (const float*)d_in[3];
    const float* b2 = (const float*)d_in[4];
    float* out = (float*)d_out;

    const size_t interBytes = (size_t)BS * IN_DIM * sizeof(float);   // 128 MB
    float* inter = (ws_size >= interBytes) ? (float*)d_ws : out;     // best-effort fallback

    dim3 grid(NBLK, BS / MT);
    dim3 blk(256);

    // layer 0: gap=1 shuffle is identity -> feature = n*64 + c
    mixer_layer_wmma<<<grid, blk, SMEM_BYTES, stream>>>(
        x, inter, w1, b1, w2, b2, /*sd=*/1, /*sn=*/64);

    // layer 1: gap=64 shuffle transposes block structure -> feature = c*64 + n
    mixer_layer_wmma<<<grid, blk, SMEM_BYTES, stream>>>(
        inter, out,
        w1 + (size_t)NBLK * BLOCKD * HID,
        b1 + (size_t)NBLK * HID,
        w2 + (size_t)NBLK * HID * BLOCKD,
        b2 + (size_t)NBLK * BLOCKD,
        /*sd=*/64, /*sn=*/1);
}
